// SubLSTM_49701361550079
// MI455X (gfx1250) — matrix operand order repaired and checked by reading, not compiled, not run on hardware
//
#include <hip/hip_runtime.h>
#include <hip/hip_bf16.h>
#include <stdint.h>

// ---------------- problem constants ----------------
#define TT 512
#define BB 32
#define HH 1024
#define II 1024
#define GG 4096   // 4*H

typedef __attribute__((ext_vector_type(16))) __bf16   bf16x16;
typedef __attribute__((ext_vector_type(8)))  float    f32x8;
typedef __attribute__((ext_vector_type(4)))  uint32_t u32x4;

union FragAB {
    bf16x16 v;
    u32x4   q[2];
};

__device__ __forceinline__ uint16_t f2bf_bits(float f) {
    union { float f; uint32_t u; } x; x.f = f;
    uint32_t r = x.u + 0x7FFFu + ((x.u >> 16) & 1u);   // round-to-nearest-even
    return (uint16_t)(r >> 16);
}

__device__ __forceinline__ float sigm(float x) {
    return 1.0f / (1.0f + __expf(-x));
}

// A fragment: 16x32 (MxK) bf16, row-major source, 16-bit A layout (ISA 7.12.2):
// lane L<16 holds row L, K {0..7,16..23}; lane L+16 holds row L, K {8..15,24..31}
__device__ __forceinline__ FragAB load_frag_A(const uint16_t* __restrict__ A,
                                              int lda, int row_base, int k, int lane) {
    FragAB f;
    const uint16_t* p = A + (size_t)(row_base + (lane & 15)) * lda + k + ((lane >> 4) << 3);
    f.q[0] = *(const u32x4*)p;
    f.q[1] = *(const u32x4*)(p + 16);
    return f;
}

// B fragment: 32x16 (KxN); source W is [N,K] row-major so a B column is a contiguous W row.
// lane n<16 holds col n, K {0..7,16..23}; lane n+16 holds col n, K {8..15,24..31}
__device__ __forceinline__ FragAB load_frag_B(const uint16_t* __restrict__ W,
                                              int ldb, int col_base, int k, int lane) {
    FragAB f;
    const uint16_t* p = W + (size_t)(col_base + (lane & 15)) * ldb + k + ((lane >> 4) << 3);
    f.q[0] = *(const u32x4*)p;
    f.q[1] = *(const u32x4*)(p + 16);
    return f;
}

#define WMMA_BF16(a, b, c) \
    __builtin_amdgcn_wmma_f32_16x16x32_bf16(false, (a), false, (b), (short)0, (c), false, false)

// ---------------- fp32 -> bf16 cast ----------------
__global__ __launch_bounds__(256) void cast_f32_bf16(const float* __restrict__ src,
                                                     uint16_t* __restrict__ dst, int n) {
    int i = blockIdx.x * 256 + threadIdx.x;
    if (i < n) dst[i] = f2bf_bits(src[i]);
}

// ---------------- batched input projection: Out[M,4096] = A[M,1024] * W[4096,1024]^T + bias
// one wave per 32(M) x 64(N) tile
__global__ __launch_bounds__(32) void gemm_xw_bias(const uint16_t* __restrict__ A,
                                                   const uint16_t* __restrict__ W,
                                                   const float* __restrict__ bias,
                                                   float* __restrict__ Out) {
    const int lane = threadIdx.x;
    const int nb = blockIdx.x * 64;
    const int mb = blockIdx.y * 32;

    f32x8 acc[2][4];
#pragma unroll
    for (int mt = 0; mt < 2; ++mt)
#pragma unroll
        for (int j = 0; j < 4; ++j) acc[mt][j] = (f32x8)0.0f;

    for (int k = 0; k < II; k += 32) {
        FragAB a0 = load_frag_A(A, II, mb,      k, lane);
        FragAB a1 = load_frag_A(A, II, mb + 16, k, lane);
#pragma unroll
        for (int j = 0; j < 4; ++j) {
            FragAB b = load_frag_B(W, II, nb + 16 * j, k, lane);
            acc[0][j] = WMMA_BF16(a0.v, b.v, acc[0][j]);
            acc[1][j] = WMMA_BF16(a1.v, b.v, acc[1][j]);
        }
    }

    const int nlo = lane & 15;
    const int mhi = (lane >> 4) * 8;
#pragma unroll
    for (int mt = 0; mt < 2; ++mt) {
#pragma unroll
        for (int j = 0; j < 4; ++j) {
            int n = nb + 16 * j + nlo;
            float bv = bias[n];
#pragma unroll
            for (int r = 0; r < 8; ++r) {
                int m = mb + mt * 16 + mhi + r;
                Out[(size_t)m * GG + n] = acc[mt][j][r] + bv;
            }
        }
    }
}

// ---------------- fused recurrent step -----------------
// gates = sigmoid(xg_t + h_prev * R^T + bh); i,o,z,f = split(gates)
// c = c*f + z - i ; h = sigmoid(c) - o
// one wave handles all 32 batch rows x 16 j-columns x all 4 gates
__global__ __launch_bounds__(32) void lstm_step(const float* __restrict__ xg_t,      // [32,4096]
                                                const uint16_t* __restrict__ hprev,  // [32,1024] bf16
                                                const uint16_t* __restrict__ R,      // [4096,1024] bf16
                                                const float* __restrict__ bh,        // [4096]
                                                float* __restrict__ c_state,         // [32,1024]
                                                float* __restrict__ h_state,         // [32,1024]
                                                uint16_t* __restrict__ h_bf16_out,   // [32,1024]
                                                float* __restrict__ out_f32)         // [32,1024] or null
{
    const int lane = threadIdx.x;
    const int jb = blockIdx.x * 16;

    f32x8 acc[4][2];   // [gate][mtile]
#pragma unroll
    for (int g = 0; g < 4; ++g)
#pragma unroll
        for (int mt = 0; mt < 2; ++mt) acc[g][mt] = (f32x8)0.0f;

    for (int k = 0; k < HH; k += 32) {
        FragAB a0 = load_frag_A(hprev, HH, 0,  k, lane);
        FragAB a1 = load_frag_A(hprev, HH, 16, k, lane);
#pragma unroll
        for (int g = 0; g < 4; ++g) {
            FragAB b = load_frag_B(R, HH, g * HH + jb, k, lane);
            acc[g][0] = WMMA_BF16(a0.v, b.v, acc[g][0]);
            acc[g][1] = WMMA_BF16(a1.v, b.v, acc[g][1]);
        }
    }

    const int j = jb + (lane & 15);
    const int mhi = (lane >> 4) * 8;
    float bhv[4];
#pragma unroll
    for (int g = 0; g < 4; ++g) bhv[g] = bh[g * HH + j];

#pragma unroll
    for (int mt = 0; mt < 2; ++mt) {
#pragma unroll
        for (int r = 0; r < 8; ++r) {
            int m = mt * 16 + mhi + r;
            size_t o4 = (size_t)m * GG + j;
            size_t o1 = (size_t)m * HH + j;
            float gi = sigm(acc[0][mt][r] + xg_t[o4]            + bhv[0]);
            float go = sigm(acc[1][mt][r] + xg_t[o4 + 1 * HH]   + bhv[1]);
            float gz = sigm(acc[2][mt][r] + xg_t[o4 + 2 * HH]   + bhv[2]);
            float gf = sigm(acc[3][mt][r] + xg_t[o4 + 3 * HH]   + bhv[3]);
            float c = c_state[o1] * gf + gz - gi;
            c_state[o1] = c;
            float h = sigm(c) - go;
            h_state[o1] = h;
            h_bf16_out[o1] = f2bf_bits(h);
            if (out_f32) out_f32[o1] = h;
        }
    }
}

// ---------------- host side ----------------
extern "C" void kernel_launch(void* const* d_in, const int* in_sizes, int n_in,
                              void* d_out, int out_size, void* d_ws, size_t ws_size,
                              hipStream_t stream) {
    const float* x   = (const float*)d_in[0];
    const float* h0  = (const float*)d_in[1];   // [2,32,1024]
    const float* c0  = (const float*)d_in[2];   // [2,32,1024]
    const float* W0  = (const float*)d_in[3];
    const float* R0  = (const float*)d_in[4];
    const float* bi0 = (const float*)d_in[5];
    const float* bh0 = (const float*)d_in[6];
    const float* W1  = (const float*)d_in[7];
    const float* R1  = (const float*)d_in[8];
    const float* bi1 = (const float*)d_in[9];
    const float* bh1 = (const float*)d_in[10];
    float* out = (float*)d_out;

    const size_t TB   = (size_t)TT * BB;          // 16384
    const size_t BH   = (size_t)BB * HH;          // 32768
    const size_t TBH  = TB * HH;                  // 16,777,216
    const size_t TBG  = TB * GG;
    const size_t WSZ  = (size_t)GG * HH;          // weight elems

    // workspace carve (all 256B aligned)
    char* p = (char*)d_ws;
    auto carve = [&](size_t bytes) { char* r = p; p += (bytes + 255) & ~(size_t)255; return r; };
    float*    xg    = (float*)   carve(TBG * 4);       // [T*B, 4096] fp32
    uint16_t* xb    = (uint16_t*)carve(TBH * 2);       // x in bf16
    uint16_t* y1b   = (uint16_t*)carve(TBH * 2);       // layer1 output seq, bf16
    uint16_t* W0b   = (uint16_t*)carve(WSZ * 2);
    uint16_t* R0b   = (uint16_t*)carve(WSZ * 2);
    uint16_t* W1b   = (uint16_t*)carve(WSZ * 2);
    uint16_t* R1b   = (uint16_t*)carve(WSZ * 2);
    uint16_t* h0b0  = (uint16_t*)carve(BH * 2);        // initial h bf16, layer 0
    uint16_t* h0b1  = (uint16_t*)carve(BH * 2);        // initial h bf16, layer 1
    uint16_t* hpp0  = (uint16_t*)carve(BH * 2);        // layer2 h ping-pong
    uint16_t* hpp1  = (uint16_t*)carve(BH * 2);
    float*    cst0  = (float*)   carve(BH * 4);
    float*    cst1  = (float*)   carve(BH * 4);
    float*    hst0  = (float*)   carve(BH * 4);
    float*    hst1  = (float*)   carve(BH * 4);
    (void)ws_size; (void)n_in; (void)in_sizes; (void)out_size;

    // ---- casts to bf16 ----
    cast_f32_bf16<<<dim3((unsigned)(TBH / 256)), dim3(256), 0, stream>>>(x, xb, (int)TBH);
    cast_f32_bf16<<<dim3((unsigned)(WSZ / 256)), dim3(256), 0, stream>>>(W0, W0b, (int)WSZ);
    cast_f32_bf16<<<dim3((unsigned)(WSZ / 256)), dim3(256), 0, stream>>>(R0, R0b, (int)WSZ);
    cast_f32_bf16<<<dim3((unsigned)(WSZ / 256)), dim3(256), 0, stream>>>(W1, W1b, (int)WSZ);
    cast_f32_bf16<<<dim3((unsigned)(WSZ / 256)), dim3(256), 0, stream>>>(R1, R1b, (int)WSZ);
    cast_f32_bf16<<<dim3((unsigned)(BH / 256)), dim3(256), 0, stream>>>(h0,      h0b0, (int)BH);
    cast_f32_bf16<<<dim3((unsigned)(BH / 256)), dim3(256), 0, stream>>>(h0 + BH, h0b1, (int)BH);

    // ---- init fp32 states ----
    hipMemcpyAsync(cst0, c0,      BH * 4, hipMemcpyDeviceToDevice, stream);
    hipMemcpyAsync(cst1, c0 + BH, BH * 4, hipMemcpyDeviceToDevice, stream);
    hipMemcpyAsync(hst0, h0,      BH * 4, hipMemcpyDeviceToDevice, stream);
    hipMemcpyAsync(hst1, h0 + BH, BH * 4, hipMemcpyDeviceToDevice, stream);

    const dim3 ggrid(GG / 64, (unsigned)(TB / 32));
    const dim3 wblk(32);

    // ================= layer 0 =================
    gemm_xw_bias<<<ggrid, wblk, 0, stream>>>(xb, W0b, bi0, xg);
    for (int t = 0; t < TT; ++t) {
        const uint16_t* hin = (t == 0) ? h0b0 : (y1b + (size_t)(t - 1) * BH);
        lstm_step<<<dim3(HH / 16), wblk, 0, stream>>>(
            xg + (size_t)t * BB * GG, hin, R0b, bh0,
            cst0, hst0, y1b + (size_t)t * BH, nullptr);
    }

    // ================= layer 1 =================
    gemm_xw_bias<<<ggrid, wblk, 0, stream>>>(y1b, W1b, bi1, xg);
    for (int t = 0; t < TT; ++t) {
        const uint16_t* hin = (t == 0) ? h0b1 : ((t & 1) ? hpp0 : hpp1);
        uint16_t* hout = (t & 1) ? hpp1 : hpp0;
        lstm_step<<<dim3(HH / 16), wblk, 0, stream>>>(
            xg + (size_t)t * BB * GG, hin, R1b, bh1,
            cst1, hst1, hout, out + (size_t)t * BH);
    }

    // ---- final (hT, cT) stacks into d_out tail ----
    float* tail = out + TBH;
    hipMemcpyAsync(tail,          hst0, BH * 4, hipMemcpyDeviceToDevice, stream);
    hipMemcpyAsync(tail + BH,     hst1, BH * 4, hipMemcpyDeviceToDevice, stream);
    hipMemcpyAsync(tail + 2 * BH, cst0, BH * 4, hipMemcpyDeviceToDevice, stream);
    hipMemcpyAsync(tail + 3 * BH, cst1, BH * 4, hipMemcpyDeviceToDevice, stream);
}
